// MultiHeadSelfAttention_54597624267251
// MI455X (gfx1250) — matrix operand order repaired and checked by reading, not compiled, not run on hardware
//
#include <hip/hip_runtime.h>
#include <cstdint>
#include <cstddef>

// ---------------------------------------------------------------------------
// Types for CDNA5 WMMA (wave32). bf16 buffers are handled as raw u16 with
// explicit RNE packing; v16bf only appears as the WMMA operand type.
// ---------------------------------------------------------------------------
typedef __bf16 v16bf __attribute__((ext_vector_type(16)));
typedef float  v8f   __attribute__((ext_vector_type(8)));
typedef unsigned int u32x4 __attribute__((ext_vector_type(4)));
typedef unsigned short u16;

union FragBF {
  v16bf v;
  u16   u[16];
  u32x4 q[2];
};

#define D_MODEL 512
#define NH      8
#define DH      256
#define BATCH   8
#define SEQ     2048
#define NG      512                 // grouped sequence length N = T/GS
#define RS      2048                // H*DH = elements per grouped row
#define BHD     (SEQ * D_MODEL)     // per-batch elements (1048576)

__device__ __forceinline__ u16 f2bf(float x) {
  unsigned int b = __float_as_uint(x);
  unsigned int lsb = (b >> 16) & 1u;
  b += 0x7fffu + lsb;               // round-to-nearest-even
  return (u16)(b >> 16);
}

__device__ __forceinline__ v8f wmma_bf16(v16bf a, v16bf b, v8f c) {
  // D = A(16x32 bf16) * B(32x16 bf16) + C(16x16 f32)
  return __builtin_amdgcn_wmma_f32_16x16x32_bf16(false, a, false, b, (short)0, c,
                                                 false, false);
}

// A fragment, 16x32 bf16: row = lane&15; K runs [8*half,8*half+8) and
// [16+8*half, 16+8*half+8). bf16 source: two 16-byte loads.
__device__ __forceinline__ v16bf load_afrag(const u16* A, int lda, int row,
                                            int k0, int lane) {
  int half = lane >> 4;
  const u16* p = A + (size_t)row * lda + k0 + 8 * half;
  FragBF f;
  f.q[0] = *(const u32x4*)(p);
  f.q[1] = *(const u32x4*)(p + 16);
  return f.v;
}

// A fragment from f32 source (projection GEMMs): convert on the fly.
__device__ __forceinline__ v16bf load_afrag(const float* A, int lda, int row,
                                            int k0, int lane) {
  int half = lane >> 4;
  const float* p = A + (size_t)row * lda + k0 + 8 * half;
  FragBF f;
#pragma unroll
  for (int i = 0; i < 8; ++i) {
    f.u[i]     = f2bf(p[i]);
    f.u[8 + i] = f2bf(p[16 + i]);
  }
  return f.v;
}

// B fragment, 32x16 bf16, source stored as rows of K ("B^T"-style, i.e.
// Bsrc[col][k]): col = lane&15, K runs [8*half,8*half+8) and [16+8*half,16+8*half+8).
__device__ __forceinline__ v16bf load_bfrag_rows(const u16* B, int ldb, int col,
                                                 int k0, int lane) {
  int half = lane >> 4;
  const u16* p = B + (size_t)col * ldb + k0 + 8 * half;
  FragBF f;
  f.q[0] = *(const u32x4*)(p);
  f.q[1] = *(const u32x4*)(p + 16);
  return f.v;
}

// ---------------------------------------------------------------------------
// Software-pipelined GEMM core: wave computes a 32x64 tile of C.
//   - 2 A fragments (rows [base,base+16) and [base+16,base+32))
//   - 4 B fragments (cols colBase + 16t)
//   - next K-step's 12 b128 loads are issued before the current 8 WMMAs,
//     so WMMA execution overlaps in-flight loads (partial loadcnt waits).
// K is compile-time -> fully unrolled pipeline.
// ---------------------------------------------------------------------------
template <int K, typename AT>
__device__ __forceinline__ void gemm_core(const AT* __restrict__ Ab, int lda,
                                          const u16* __restrict__ Bb, int ldb,
                                          int rowWaveBase, int colBase, int lane,
                                          v8f* acc) {
  const int cl = lane & 15;
  const int row0 = rowWaveBase + cl;
  v16bf a0 = load_afrag(Ab, lda, row0, 0, lane);
  v16bf a1 = load_afrag(Ab, lda, row0 + 16, 0, lane);
  v16bf b0 = load_bfrag_rows(Bb, ldb, colBase + cl, 0, lane);
  v16bf b1 = load_bfrag_rows(Bb, ldb, colBase + 16 + cl, 0, lane);
  v16bf b2 = load_bfrag_rows(Bb, ldb, colBase + 32 + cl, 0, lane);
  v16bf b3 = load_bfrag_rows(Bb, ldb, colBase + 48 + cl, 0, lane);
#pragma unroll
  for (int kb = 0; kb < K; kb += 32) {
    const int nkb = (kb + 32 < K) ? (kb + 32) : kb;   // last iter: harmless reload
    v16bf na0 = load_afrag(Ab, lda, row0, nkb, lane);
    v16bf na1 = load_afrag(Ab, lda, row0 + 16, nkb, lane);
    v16bf nb0 = load_bfrag_rows(Bb, ldb, colBase + cl, nkb, lane);
    v16bf nb1 = load_bfrag_rows(Bb, ldb, colBase + 16 + cl, nkb, lane);
    v16bf nb2 = load_bfrag_rows(Bb, ldb, colBase + 32 + cl, nkb, lane);
    v16bf nb3 = load_bfrag_rows(Bb, ldb, colBase + 48 + cl, nkb, lane);
    acc[0] = wmma_bf16(a0, b0, acc[0]);
    acc[1] = wmma_bf16(a0, b1, acc[1]);
    acc[2] = wmma_bf16(a0, b2, acc[2]);
    acc[3] = wmma_bf16(a0, b3, acc[3]);
    acc[4] = wmma_bf16(a1, b0, acc[4]);
    acc[5] = wmma_bf16(a1, b1, acc[5]);
    acc[6] = wmma_bf16(a1, b2, acc[6]);
    acc[7] = wmma_bf16(a1, b3, acc[7]);
    a0 = na0; a1 = na1; b0 = nb0; b1 = nb1; b2 = nb2; b3 = nb3;
  }
}

// ---------------------------------------------------------------------------
// Prep kernels
// ---------------------------------------------------------------------------

// W[512][512] f32 -> Wt[n][k] bf16 (transposed, so GEMM B-loads are contiguous)
__global__ void k_transpose_w(const float* __restrict__ W, u16* __restrict__ Wt) {
  int idx = blockIdx.x * 256 + threadIdx.x;      // 262144 elements
  int n = idx >> 9, k = idx & 511;
  Wt[idx] = f2bf(W[(size_t)k * 512 + n]);
}

// Relative sinusoidal PE slice PE[2:4094] (4092 rows), padded to 4096 rows of 0.
__global__ void k_pe(u16* __restrict__ PEbf) {
  int idx = blockIdx.x * 256 + threadIdx.x;      // 4096*512 elements
  int p = idx >> 9, c = idx & 511;
  float val = 0.0f;
  if (p < 4092) {
    int pidx = p + 2;
    float pos = (pidx < 2048) ? (float)(2047 - pidx) : (float)(2048 - pidx);
    int s = c >> 1;
    float freq = __expf(-(float)s * (9.210340371976184f / 256.0f)); // 10000^(-s/256)
    float ang = pos * freq;
    val = (c & 1) ? __cosf(ang) : __sinf(ang);
  }
  PEbf[idx] = f2bf(val);
}

// Per-head transpose of projected V: Vt[(b*8+h)*256+dh][j] = Vh[b,h][j][dh],
// so the attention-output GEMM's B fragments are contiguous b128 loads.
__global__ void k_transpose_v(const u16* __restrict__ Vp, u16* __restrict__ Vt) {
  int idx = blockIdx.x * 256 + threadIdx.x;      // 8388608 elements
  int j = idx & 511;
  int rest = idx >> 9;
  int dh = rest & 255;
  int zz = rest >> 8;                            // b*8+h
  int b = zz >> 3, h = zz & 7;
  Vt[idx] = Vp[(size_t)b * BHD + (size_t)j * RS + (size_t)h * DH + dh];
}

// ---------------------------------------------------------------------------
// Generic K=512, N=512 GEMM: C = A @ Wt^T + bias, with epilogue variants.
//   O0 (bf16) = C + add0[n], O1 (bf16) = C + add1[n], Of (f32) = C
// Block: 128 threads (4 waves); block tile 128x64; wave tile 32x64.
// gridDim = (M/128, 8).
// ---------------------------------------------------------------------------
template <typename AT>
__global__ void k_gemm512(const AT* __restrict__ A, const u16* __restrict__ Wt,
                          const float* __restrict__ bias,
                          const float* __restrict__ add0,
                          const float* __restrict__ add1,
                          u16* __restrict__ O0, u16* __restrict__ O1,
                          float* __restrict__ Of) {
  int lane = threadIdx.x & 31;
  int wave = threadIdx.x >> 5;
  int rowBase = blockIdx.x * 128 + wave * 32;
  int colBase = blockIdx.y * 64;
  int half = lane >> 4;

  v8f acc[8] = {};
  gemm_core<512>(A, 512, Wt, 512, rowBase, colBase, lane, acc);

#pragma unroll
  for (int g = 0; g < 2; ++g) {
#pragma unroll
    for (int t = 0; t < 4; ++t) {
      int col = colBase + 16 * t + (lane & 15);
      float bsum = bias[col];
      float a0 = add0 ? add0[col] : 0.0f;
      float a1 = add1 ? add1[col] : 0.0f;
#pragma unroll
      for (int r = 0; r < 8; ++r) {
        int rr = rowBase + g * 16 + half * 8 + r;
        float c = acc[g * 4 + t][r] + bsum;
        size_t idx = (size_t)rr * 512 + col;
        if (O0) O0[idx] = f2bf(c + a0);
        if (O1) O1[idx] = f2bf(c + a1);
        if (Of) Of[idx] = c;
      }
    }
  }
}

// ---------------------------------------------------------------------------
// Relative-position scores: rel[b,h,i,m] = Qv_head[i]·Eh[m], fused rel_to_abs:
// scatter rel element (i,m) to S[b,h, i, j=m-511+i] (bijection; no init needed).
// gridDim = (4 i-tiles, 16 m-tiles, 64 bh).
// ---------------------------------------------------------------------------
__global__ void k_scores_e(const u16* __restrict__ Qv, const u16* __restrict__ E,
                           float* __restrict__ S) {
  int z = blockIdx.z, b = z >> 3, h = z & 7;
  int lane = threadIdx.x & 31, wave = threadIdx.x >> 5;
  int rowBase = blockIdx.x * 128 + wave * 32;   // query index i
  int colBase = blockIdx.y * 64;                // relative index m
  int half = lane >> 4;

  v8f acc[8] = {};
  gemm_core<DH>(Qv + (size_t)b * BHD + h * DH, RS, E + h * DH, RS,
                rowBase, colBase, lane, acc);

  float* Sz = S + (size_t)z * NG * NG;
#pragma unroll
  for (int g = 0; g < 2; ++g) {
#pragma unroll
    for (int t = 0; t < 4; ++t) {
      int m = colBase + 16 * t + (lane & 15);
#pragma unroll
      for (int r = 0; r < 8; ++r) {
        int i = rowBase + g * 16 + half * 8 + r;
        int j = m - (NG - 1) + i;               // rel_to_abs skew
        if (j >= 0 && j < NG) Sz[(size_t)i * NG + j] = acc[g * 4 + t][r];
      }
    }
  }
}

// ---------------------------------------------------------------------------
// Content scores: S = (S + Qu_head·Kh^T) / sqrt(DH), masked.
// gridDim = (4, 8, 64).
// ---------------------------------------------------------------------------
__global__ void k_scores_k(const u16* __restrict__ Qu, const u16* __restrict__ Kp,
                           const unsigned char* __restrict__ mask,
                           float* __restrict__ S) {
  int z = blockIdx.z, b = z >> 3, h = z & 7;
  int lane = threadIdx.x & 31, wave = threadIdx.x >> 5;
  int rowBase = blockIdx.x * 128 + wave * 32;   // i
  int colBase = blockIdx.y * 64;                // j
  int half = lane >> 4;

  v8f acc[8] = {};
  gemm_core<DH>(Qu + (size_t)b * BHD + h * DH, RS, Kp + (size_t)b * BHD + h * DH, RS,
                rowBase, colBase, lane, acc);

  float* Sz = S + (size_t)z * NG * NG;
#pragma unroll
  for (int g = 0; g < 2; ++g) {
#pragma unroll
    for (int t = 0; t < 4; ++t) {
      int j = colBase + 16 * t + (lane & 15);
#pragma unroll
      for (int r = 0; r < 8; ++r) {
        int i = rowBase + g * 16 + half * 8 + r;
        size_t idx = (size_t)i * NG + j;
        float val = (Sz[idx] + acc[g * 4 + t][r]) * 0.0625f;   // 1/sqrt(256)
        if (mask[(size_t)b * SEQ * SEQ + (size_t)(i * 4) * SEQ + (size_t)(j * 4)])
          val = -3.402823466e38f;
        Sz[idx] = val;
      }
    }
  }
}

// ---------------------------------------------------------------------------
// Row softmax over 512 keys; output probabilities as bf16.
// gridDim = (512, 64); 128 threads.
// ---------------------------------------------------------------------------
__global__ void k_softmax(const float* __restrict__ S, u16* __restrict__ P) {
  int i = blockIdx.x, z = blockIdx.y, t = threadIdx.x;
  const float* row = S + ((size_t)z * NG + i) * NG;
  u16* prow = P + ((size_t)z * NG + i) * NG;
  __shared__ float red[128];

  float x[4];
  float mx = -3.402823466e38f;
#pragma unroll
  for (int c = 0; c < 4; ++c) { x[c] = row[t + c * 128]; mx = fmaxf(mx, x[c]); }
  red[t] = mx; __syncthreads();
  for (int s = 64; s > 0; s >>= 1) {
    if (t < s) red[t] = fmaxf(red[t], red[t + s]);
    __syncthreads();
  }
  mx = red[0]; __syncthreads();

  float sum = 0.0f;
#pragma unroll
  for (int c = 0; c < 4; ++c) { x[c] = __expf(x[c] - mx); sum += x[c]; }
  red[t] = sum; __syncthreads();
  for (int s = 64; s > 0; s >>= 1) {
    if (t < s) red[t] += red[t + s];
    __syncthreads();
  }
  float inv = 1.0f / red[0];
#pragma unroll
  for (int c = 0; c < 4; ++c) prow[t + c * 128] = f2bf(x[c] * inv);
}

// ---------------------------------------------------------------------------
// out_head = P @ V_head (K=512, N=DH=256), B from per-head transposed Vt so
// all operand traffic is contiguous b128. Writes bf16 into the attn buffer at
// the transpose(0,2,1,3).reshape(B,T,D) flat layout.
// gridDim = (4, 4, 64).
// ---------------------------------------------------------------------------
__global__ void k_outgemm(const u16* __restrict__ P, const u16* __restrict__ Vt,
                          u16* __restrict__ attn) {
  int z = blockIdx.z, b = z >> 3, h = z & 7;
  int lane = threadIdx.x & 31, wave = threadIdx.x >> 5;
  int rowBase = blockIdx.x * 128 + wave * 32;   // n
  int colBase = blockIdx.y * 64;                // dh
  int half = lane >> 4;

  v8f acc[8] = {};
  gemm_core<NG>(P + (size_t)z * NG * NG, NG, Vt + (size_t)z * DH * NG, NG,
                rowBase, colBase, lane, acc);

#pragma unroll
  for (int g = 0; g < 2; ++g) {
#pragma unroll
    for (int t = 0; t < 4; ++t) {
      int dh = colBase + 16 * t + (lane & 15);
#pragma unroll
      for (int r = 0; r < 8; ++r) {
        int n = rowBase + g * 16 + half * 8 + r;
        attn[(size_t)b * BHD + (size_t)n * RS + (size_t)h * DH + dh] =
            f2bf(acc[g * 4 + t][r]);
      }
    }
  }
}

// ---------------------------------------------------------------------------
// Host launcher
// ---------------------------------------------------------------------------
extern "C" void kernel_launch(void* const* d_in, const int* in_sizes, int n_in,
                              void* d_out, int out_size, void* d_ws, size_t ws_size,
                              hipStream_t stream) {
  (void)in_sizes; (void)n_in; (void)out_size; (void)ws_size;
  const float* Q  = (const float*)d_in[0];
  const float* K  = (const float*)d_in[1];
  const float* V  = (const float*)d_in[2];
  const unsigned char* mask = (const unsigned char*)d_in[3];
  const float* Wq = (const float*)d_in[4];
  const float* bq = (const float*)d_in[5];
  const float* Wk = (const float*)d_in[6];
  const float* bk = (const float*)d_in[7];
  const float* Wv = (const float*)d_in[8];
  const float* bv = (const float*)d_in[9];
  const float* Wp = (const float*)d_in[10];
  const float* bp = (const float*)d_in[11];
  const float* Wo = (const float*)d_in[12];
  const float* bo = (const float*)d_in[13];
  const float* u  = (const float*)d_in[14];
  const float* v  = (const float*)d_in[15];
  float* out = (float*)d_out;
  char* ws = (char*)d_ws;

  // Workspace layout (all 256-B aligned offsets)
  u16*   wqt  = (u16*)(ws + 0);           // 5 transposed bf16 weights, 512KB each
  u16*   wkt  = (u16*)(ws + 524288);
  u16*   wvt  = (u16*)(ws + 1048576);
  u16*   wpt  = (u16*)(ws + 1572864);
  u16*   wot  = (u16*)(ws + 2097152);
  u16*   pebf = (u16*)(ws + 2621440);     // 4096x512 bf16 (padded PE slice)
  u16*   ebf  = (u16*)(ws + 6815744);     // 4096x512 bf16 (projected E)
  u16*   qu   = (u16*)(ws + 11010048);    // 16384x512 bf16
  u16*   qv   = (u16*)(ws + 27787264);
  u16*   kp   = (u16*)(ws + 44564480);
  u16*   vp   = (u16*)(ws + 61341696);
  float* S    = (float*)(ws + 78118912);  // 64x512x512 f32 scores
  u16*   P    = (u16*)(ws + 145227776);   // 64x512x512 bf16 probs
  u16*   attn = (u16*)(ws + 178782208);   // 16384x512 bf16
  u16*   vt   = (u16*)(ws + 195559424);   // per-head transposed V, 16MB

  // Stage 0: weight transpose+convert, PE table
  k_transpose_w<<<1024, 256, 0, stream>>>(Wq, wqt);
  k_transpose_w<<<1024, 256, 0, stream>>>(Wk, wkt);
  k_transpose_w<<<1024, 256, 0, stream>>>(Wv, wvt);
  k_transpose_w<<<1024, 256, 0, stream>>>(Wp, wpt);
  k_transpose_w<<<1024, 256, 0, stream>>>(Wo, wot);
  k_pe<<<8192, 256, 0, stream>>>(pebf);

  // Stage 1: projections (WMMA bf16, f32 accumulate)
  k_gemm512<float><<<dim3(128, 8), 128, 0, stream>>>(Q, wqt, bq, u, v, qu, qv, nullptr);
  k_gemm512<float><<<dim3(128, 8), 128, 0, stream>>>(K, wkt, bk, nullptr, nullptr, kp, nullptr, nullptr);
  k_gemm512<float><<<dim3(128, 8), 128, 0, stream>>>(V, wvt, bv, nullptr, nullptr, vp, nullptr, nullptr);
  k_gemm512<u16><<<dim3(32, 8), 128, 0, stream>>>(pebf, wpt, bp, nullptr, nullptr, ebf, nullptr, nullptr);
  k_transpose_v<<<32768, 256, 0, stream>>>(vp, vt);

  // Stage 2: scores (rel GEMM with fused rel_to_abs scatter, then content GEMM)
  k_scores_e<<<dim3(4, 16, 64), 128, 0, stream>>>(qv, ebf, S);
  k_scores_k<<<dim3(4, 8, 64), 128, 0, stream>>>(qu, kp, mask, S);

  // Stage 3: softmax -> bf16 probabilities
  k_softmax<<<dim3(512, 64), 128, 0, stream>>>(S, P);

  // Stage 4: attention output, then final projection to f32 d_out
  k_outgemm<<<dim3(4, 4, 64), 128, 0, stream>>>(P, vt, attn);
  k_gemm512<u16><<<dim3(128, 8), 128, 0, stream>>>(attn, wot, bo, nullptr, nullptr, nullptr, nullptr, out);
}
